// LSTMDecoder_49624052138309
// MI455X (gfx1250) — matrix operand order repaired and checked by reading, not compiled, not run on hardware
//
#include <hip/hip_runtime.h>
#include <hip/hip_bf16.h>
#include <math.h>

typedef __attribute__((ext_vector_type(16))) _Float16 v16h;
typedef __attribute__((ext_vector_type(8)))  _Float16 v8h;
typedef __attribute__((ext_vector_type(8)))  float    v8f;

#define HDIM   128
#define LDP    136        // LDS row pitch in f16 elems (272B -> bank-conflict-free)
#define BM     64         // batch rows per block
#define BATCH  65536
#define NUM_LAW  103
#define NUM_ACCU 119
#define NUM_TIME 11

// f16 transposed-weight workspace layout (element offsets)
constexpr int OFF_WXT = 0;                         // 3 x [512][128]
constexpr int OFF_WHT = OFF_WXT + 3 * 512 * 128;   // 3 x [512][128]
constexpr int OFF_HFC = OFF_WHT + 3 * 512 * 128;   // [128][128]
constexpr int OFF_CFC = OFF_HFC + 128 * 128;       // [128][128]
constexpr int OFF_MID = OFF_CFC + 128 * 128;       // 3 x [128][128]
constexpr int OFF_O1  = OFF_MID + 3 * 128 * 128;   // [112][128] (103 padded)
constexpr int OFF_O2  = OFF_O1  + 112 * 128;       // [128][128] (119 padded)
constexpr int OFF_O3  = OFF_O2  + 128 * 128;       // [16][128]  (11 padded)
constexpr int WS_ELEMS = OFF_O3 + 16 * 128;        // = 507904 f16 (~0.97 MB)

// ---------------------------------------------------------------------------
// Fast transcendentals: CDNA5 has a hardware V_TANH_F32 (TRANS pipe, which
// co-executes with the 16-bit WMMA/XDL pipe). sigmoid(x)=0.5*tanh(x/2)+0.5
// keeps the whole gate nonlinearity on the trans pipe with no divides.
// ---------------------------------------------------------------------------
#if __has_builtin(__builtin_amdgcn_tanhf)
__device__ __forceinline__ float fast_tanh(float x) {
  return __builtin_amdgcn_tanhf(x);
}
__device__ __forceinline__ float sigm(float x) {
  return __builtin_fmaf(0.5f, __builtin_amdgcn_tanhf(0.5f * x), 0.5f);
}
#else
__device__ __forceinline__ float fast_tanh(float x) { return tanhf(x); }
__device__ __forceinline__ float sigm(float x) {
  // rcp-based: avoids the IEEE divide (v_div_fixup) chain
  return __builtin_amdgcn_rcpf(1.0f + __expf(-x));
}
#endif

// ---------------------------------------------------------------------------
// Prep: convert fp32 weights to f16, transposed to [N][K] row-major, pad heads
// ---------------------------------------------------------------------------
__global__ void prep_weights(const float* __restrict__ Wx,
                             const float* __restrict__ Wh,
                             const float* __restrict__ hfcW,
                             const float* __restrict__ cfcW,
                             const float* __restrict__ midW,
                             const float* __restrict__ o1W,
                             const float* __restrict__ o2W,
                             const float* __restrict__ o3W,
                             _Float16* __restrict__ ws) {
  int idx = blockIdx.x * 256 + threadIdx.x;
  if (idx >= WS_ELEMS) return;
  float v;
  if (idx < OFF_WHT) {                     // WxT: t, n(0..511), k(0..127)
    int r = idx - OFF_WXT, t = r / 65536, q = r % 65536, n = q / 128, k = q % 128;
    v = Wx[t * 65536 + k * 512 + n];
  } else if (idx < OFF_HFC) {              // WhT
    int r = idx - OFF_WHT, t = r / 65536, q = r % 65536, n = q / 128, k = q % 128;
    v = Wh[t * 65536 + k * 512 + n];
  } else if (idx < OFF_CFC) {              // hfcT
    int r = idx - OFF_HFC, n = r / 128, k = r % 128;
    v = hfcW[k * 128 + n];
  } else if (idx < OFF_MID) {              // cfcT
    int r = idx - OFF_CFC, n = r / 128, k = r % 128;
    v = cfcW[k * 128 + n];
  } else if (idx < OFF_O1) {               // midT
    int r = idx - OFF_MID, t = r / 16384, q = r % 16384, n = q / 128, k = q % 128;
    v = midW[t * 16384 + k * 128 + n];
  } else if (idx < OFF_O2) {               // out1T padded 112
    int r = idx - OFF_O1, n = r / 128, k = r % 128;
    v = (n < NUM_LAW) ? o1W[k * NUM_LAW + n] : 0.0f;
  } else if (idx < OFF_O3) {               // out2T padded 128
    int r = idx - OFF_O2, n = r / 128, k = r % 128;
    v = (n < NUM_ACCU) ? o2W[k * NUM_ACCU + n] : 0.0f;
  } else {                                 // out3T padded 16
    int r = idx - OFF_O3, n = r / 128, k = r % 128;
    v = (n < NUM_TIME) ? o3W[k * NUM_TIME + n] : 0.0f;
  }
  ws[idx] = (_Float16)v;
}

// ---------------------------------------------------------------------------
// WMMA helpers
// ---------------------------------------------------------------------------
__device__ __forceinline__ v8f wmma32(v16h a, v16h b, v8f c) {
  return __builtin_amdgcn_wmma_f32_16x16x32_f16(false, a, false, b,
                                                (short)0, c, false, false);
}

__device__ __forceinline__ v8f vzero8() {
  v8f z = {0.f, 0.f, 0.f, 0.f, 0.f, 0.f, 0.f, 0.f};
  return z;
}

// A fragment (16x32 f16) from an LDS strip (rows 0..15 of this wave, pitch LDP)
// lanes 0-15: row M=lane, K=kb..kb+15
// lanes 16-31: row M=lane-16, K=kb+16..kb+31
__device__ __forceinline__ v16h load_a(const _Float16* __restrict__ strip,
                                       int kb, int lane) {
  int row = lane & 15;
  int kh  = (lane >> 4) * 16;
  const _Float16* p = strip + row * LDP + kb + kh;
  union { v16h v; v8h h[2]; } u;
  u.h[0] = *(const v8h*)(p);
  u.h[1] = *(const v8h*)(p + 8);
  return u.v;
}

// B fragment (32x16 f16) from f16 WT[N][128] in global (L2-resident)
// lane n (0-15): column nb+n, K=kb..kb+15 ; lanes 16-31: same col, K=kb+16..kb+31
__device__ __forceinline__ v16h load_b(const _Float16* __restrict__ wT,
                                       int nb, int kb, int lane) {
  int col = lane & 15;
  int kh  = (lane >> 4) * 16;
  const _Float16* p = wT + (nb + col) * HDIM + kb + kh;
  union { v16h v; v8h h[2]; } u;
  u.h[0] = *(const v8h*)(p);
  u.h[1] = *(const v8h*)(p + 8);
  return u.v;
}

// ---------------------------------------------------------------------------
// One LSTM cell over this wave's 16-row strip.
// z = x@Wx (+ h@Wh) + b ; gates i,f,g,o ; c_new = sig(f)*c + sig(i)*tanh(g);
// h_new = sig(o)*tanh(c_new). All activations f16 in LDS, f32 accumulation.
// ---------------------------------------------------------------------------
template <bool HAS_H, bool HAS_C, bool STORE_C>
__device__ __forceinline__ void lstm_stage(const _Float16* __restrict__ xA,
                                           const _Float16* __restrict__ hA,
                                           const _Float16* __restrict__ WxT,
                                           const _Float16* __restrict__ WhT,
                                           const float* __restrict__ bias,
                                           const _Float16* __restrict__ cprev,
                                           _Float16* __restrict__ hdst,
                                           _Float16* __restrict__ cdst,
                                           int lane) {
  for (int nt = 0; nt < 8; ++nt) {
    int nb = nt * 16;
    v8f acc[4];
#pragma unroll
    for (int g = 0; g < 4; ++g) acc[g] = vzero8();
#pragma unroll
    for (int ks = 0; ks < 4; ++ks) {
      int kb = ks * 32;
      v16h ax = load_a(xA, kb, lane);
      v16h ah;
      if (HAS_H) ah = load_a(hA, kb, lane);
#pragma unroll
      for (int g = 0; g < 4; ++g) {
        v16h bx = load_b(WxT, g * HDIM + nb, kb, lane);
        acc[g] = wmma32(ax, bx, acc[g]);
        if (HAS_H) {
          v16h bh = load_b(WhT, g * HDIM + nb, kb, lane);
          acc[g] = wmma32(ah, bh, acc[g]);
        }
      }
    }
    int n = nb + (lane & 15);
    float bi = bias[n], bf = bias[HDIM + n], bg = bias[2 * HDIM + n],
          bo = bias[3 * HDIM + n];
#pragma unroll
    for (int r = 0; r < 8; ++r) {
      int m = ((lane >> 4) << 3) + r;
      float iv = sigm(acc[0][r] + bi);
      float fv = sigm(acc[1][r] + bf);
      float gv = fast_tanh(acc[2][r] + bg);
      float ov = sigm(acc[3][r] + bo);
      float cp = HAS_C ? (float)cprev[m * LDP + n] : 0.0f;
      float cn = fv * cp + iv * gv;
      hdst[m * LDP + n] = (_Float16)(ov * fast_tanh(cn));
      if (STORE_C) cdst[m * LDP + n] = (_Float16)cn;
    }
  }
}

// ---------------------------------------------------------------------------
// Head: m = relu(h@midT + mid_b); o = m@outT + out_b -> global fp32
// ---------------------------------------------------------------------------
__device__ __forceinline__ void head_stage(const _Float16* __restrict__ hA,
                                           const _Float16* __restrict__ midT,
                                           const float* __restrict__ midb,
                                           const _Float16* __restrict__ outT,
                                           const float* __restrict__ outb,
                                           int NUM, int NT,
                                           float* __restrict__ og, int grow0,
                                           _Float16* __restrict__ ms, int lane) {
  for (int nt = 0; nt < 8; ++nt) {
    int nb = nt * 16;
    v8f acc = vzero8();
#pragma unroll
    for (int ks = 0; ks < 4; ++ks) {
      int kb = ks * 32;
      v16h a = load_a(hA, kb, lane);
      v16h b = load_b(midT, nb, kb, lane);
      acc = wmma32(a, b, acc);
    }
    int n = nb + (lane & 15);
    float bb = midb[n];
#pragma unroll
    for (int r = 0; r < 8; ++r) {
      int m = ((lane >> 4) << 3) + r;
      float v = acc[r] + bb;
      ms[m * LDP + n] = (_Float16)(v > 0.0f ? v : 0.0f);
    }
  }
  for (int nt = 0; nt < NT; ++nt) {
    int nb = nt * 16;
    v8f acc = vzero8();
#pragma unroll
    for (int ks = 0; ks < 4; ++ks) {
      int kb = ks * 32;
      v16h a = load_a(ms, kb, lane);
      v16h b = load_b(outT, nb, kb, lane);
      acc = wmma32(a, b, acc);
    }
    int n = nb + (lane & 15);
    if (n < NUM) {
      float bb = outb[n];
#pragma unroll
      for (int r = 0; r < 8; ++r) {
        int m = ((lane >> 4) << 3) + r;
        og[(size_t)(grow0 + m) * NUM + n] = acc[r] + bb;
      }
    }
  }
}

// ---------------------------------------------------------------------------
// Fully fused multi-task LSTM decoder. One block = 64 batch rows, 4 waves;
// after x staging each wave works only on its own 16-row strip (no barriers).
// ---------------------------------------------------------------------------
__global__ __launch_bounds__(128) void fused_lstm_decoder(
    const float* __restrict__ x, const float* __restrict__ b,
    const float* __restrict__ hfc_b, const float* __restrict__ cfc_b,
    const float* __restrict__ mid_b, const float* __restrict__ o1b,
    const float* __restrict__ o2b, const float* __restrict__ o3b,
    const _Float16* __restrict__ ws, float* __restrict__ out) {
  __shared__ __align__(16) _Float16 smem[7 * BM * LDP];
  _Float16* Sx   = smem;
  _Float16* Sh1  = smem + 1 * BM * LDP;
  _Float16* Sc1  = smem + 2 * BM * LDP;   // later reused as h3
  _Float16* Sh2  = smem + 3 * BM * LDP;
  _Float16* Sc2  = smem + 4 * BM * LDP;
  _Float16* Sh3i = smem + 5 * BM * LDP;
  _Float16* Sc3i = smem + 6 * BM * LDP;   // later reused as m

  int tid = threadIdx.x;
  int bm  = blockIdx.x * BM;

  // stage x tile: fp32 -> f16 LDS (coalesced)
  for (int i = tid; i < BM * HDIM; i += 128) {
    int row = i >> 7, col = i & 127;
    Sx[row * LDP + col] = (_Float16)x[(size_t)(bm + row) * HDIM + col];
  }
  __syncthreads();

  int wv = tid >> 5, lane = tid & 31;
  int sb = wv * 16;  // strip base row inside block
  const _Float16* xs  = Sx   + sb * LDP;
  _Float16* h1s = Sh1  + sb * LDP;
  _Float16* c1s = Sc1  + sb * LDP;
  _Float16* h2s = Sh2  + sb * LDP;
  _Float16* c2s = Sc2  + sb * LDP;
  _Float16* h3is = Sh3i + sb * LDP;
  _Float16* c3is = Sc3i + sb * LDP;
  _Float16* h3s = c1s;    // c1 dead after fusion stage
  _Float16* ms  = c3is;   // c3_in dead after cell 3

  // cell 1 (h_prev = c_prev = 0  =>  h@Wh term vanishes)
  lstm_stage<false, false, true>(xs, nullptr, ws + OFF_WXT, nullptr,
                                 b + 0, nullptr, h1s, c1s, lane);
  // cell 2
  lstm_stage<true, true, true>(xs, h1s, ws + OFF_WXT + 65536,
                               ws + OFF_WHT + 65536, b + 512, c1s, h2s, c2s,
                               lane);
  // fusion: h3_in = h1 + h2@hfc + hfc_b ; c3_in = c1 + c2@cfc + cfc_b
  for (int nt = 0; nt < 8; ++nt) {
    int nb = nt * 16;
    v8f aH = vzero8(), aC = vzero8();
#pragma unroll
    for (int ks = 0; ks < 4; ++ks) {
      int kb = ks * 32;
      v16h a2 = load_a(h2s, kb, lane);
      v16h bh = load_b(ws + OFF_HFC, nb, kb, lane);
      aH = wmma32(a2, bh, aH);
      v16h ac2 = load_a(c2s, kb, lane);
      v16h bc = load_b(ws + OFF_CFC, nb, kb, lane);
      aC = wmma32(ac2, bc, aC);
    }
    int n = nb + (lane & 15);
    float bh = hfc_b[n], bc = cfc_b[n];
#pragma unroll
    for (int r = 0; r < 8; ++r) {
      int m = ((lane >> 4) << 3) + r;
      h3is[m * LDP + n] = (_Float16)(aH[r] + bh + (float)h1s[m * LDP + n]);
      c3is[m * LDP + n] = (_Float16)(aC[r] + bc + (float)c1s[m * LDP + n]);
    }
  }
  // cell 3 (c3 not needed downstream -> don't store)
  lstm_stage<true, true, false>(xs, h3is, ws + OFF_WXT + 2 * 65536,
                                ws + OFF_WHT + 2 * 65536, b + 1024, c3is, h3s,
                                nullptr, lane);

  // heads (outputs concatenated flat: o1 | o2 | o3)
  int grow0 = bm + sb;
  float* o1g = out;
  float* o2g = out + (size_t)BATCH * NUM_LAW;
  float* o3g = out + (size_t)BATCH * (NUM_LAW + NUM_ACCU);
  head_stage(h1s, ws + OFF_MID,             mid_b,           ws + OFF_O1, o1b,
             NUM_LAW, 7, o1g, grow0, ms, lane);
  head_stage(h2s, ws + OFF_MID + 16384,     mid_b + HDIM,    ws + OFF_O2, o2b,
             NUM_ACCU, 8, o2g, grow0, ms, lane);
  head_stage(h3s, ws + OFF_MID + 2 * 16384, mid_b + 2 * HDIM, ws + OFF_O3, o3b,
             NUM_TIME, 1, o3g, grow0, ms, lane);
}

// ---------------------------------------------------------------------------
extern "C" void kernel_launch(void* const* d_in, const int* in_sizes, int n_in,
                              void* d_out, int out_size, void* d_ws,
                              size_t ws_size, hipStream_t stream) {
  const float* x     = (const float*)d_in[0];
  const float* Wx    = (const float*)d_in[1];
  const float* Wh    = (const float*)d_in[2];
  const float* b     = (const float*)d_in[3];
  const float* hfcW  = (const float*)d_in[4];
  const float* hfcb  = (const float*)d_in[5];
  const float* cfcW  = (const float*)d_in[6];
  const float* cfcb  = (const float*)d_in[7];
  const float* midW  = (const float*)d_in[8];
  const float* midb  = (const float*)d_in[9];
  const float* o1W   = (const float*)d_in[10];
  const float* o1b   = (const float*)d_in[11];
  const float* o2W   = (const float*)d_in[12];
  const float* o2b   = (const float*)d_in[13];
  const float* o3W   = (const float*)d_in[14];
  const float* o3b   = (const float*)d_in[15];
  _Float16* ws = (_Float16*)d_ws;  // needs WS_ELEMS*2 ~ 1 MB of scratch
  float* out = (float*)d_out;

  prep_weights<<<(WS_ELEMS + 255) / 256, 256, 0, stream>>>(
      Wx, Wh, hfcW, cfcW, midW, o1W, o2W, o3W, ws);
  fused_lstm_decoder<<<BATCH / BM, 128, 0, stream>>>(
      x, b, hfcb, cfcb, midb, o1b, o2b, o3b, ws, out);
}
